// LinearCrossAttention_89352499626420
// MI455X (gfx1250) — matrix operand order repaired and checked by reading, not compiled, not run on hardware
//
#include <hip/hip_runtime.h>
#include <hip/hip_bf16.h>

typedef __bf16 bf16;
typedef __attribute__((ext_vector_type(8)))  __bf16 v8bf;
typedef __attribute__((ext_vector_type(16))) __bf16 v16bf;
typedef __attribute__((ext_vector_type(8)))  float  v8f;
typedef __attribute__((ext_vector_type(4)))  float  f32x4;
typedef __attribute__((ext_vector_type(4)))  unsigned int u32x4;
typedef __attribute__((ext_vector_type(4)))  int    i32x4;
typedef __attribute__((ext_vector_type(8)))  int    i32x8;

union V16 { v16bf v; v8bf h[2]; };

// ---------------------------------------------------------------------------
// f32 -> bf16 cast
// ---------------------------------------------------------------------------
__global__ void cast_f32_to_bf16(const float* __restrict__ in,
                                 bf16* __restrict__ out, int n) {
  int i = blockIdx.x * blockDim.x + threadIdx.x;
  int stride = gridDim.x * blockDim.x;
  for (; i < n; i += stride) out[i] = (bf16)in[i];
}

// ---------------------------------------------------------------------------
// TDM: DMA one 128-row x 32-col bf16 tile (row stride = K elems) from global
// into LDS at byte offset lds_addr, padding each 64B row with 16B so the LDS
// row stride is 80B (= 40 bf16), matching the compute-side layout.
// Descriptor per CDNA5 ISA.
// ---------------------------------------------------------------------------
static __device__ __forceinline__ void tdm_load_tile_128x32(
    const bf16* gsrc, unsigned lds_addr, int K, int rows_total) {
  unsigned long long ga = (unsigned long long)(uintptr_t)gsrc;
  u32x4 g0;
  g0.x = 1u;                                    // count=1 (valid), user mode
  g0.y = lds_addr;                              // LDS byte address
  g0.z = (unsigned)(ga & 0xffffffffu);          // global_addr[31:0]
  g0.w = (unsigned)((ga >> 32) & 0x01ffffffu)   // global_addr[56:32]
       | (2u << 30);                            // type = 2 ("image")
  i32x8 g1;
  g1[0] = (1 << 16)      // data_size = 2 bytes
        | (1 << 20)      // pad_enable
        | (3 << 22)      // pad_interval: 16 DWORDs (one 64B tile row)
        | (3 << 25);     // pad_amount: 4 DWORDs (16B) -> 80B LDS row stride
  g1[1] = (K & 0xffff) << 16;                          // tensor_dim0[15:0]
  g1[2] = ((K >> 16) & 0xffff)                         // tensor_dim0[31:16]
        | ((rows_total & 0xffff) << 16);               // tensor_dim1[15:0]
  g1[3] = ((rows_total >> 16) & 0xffff)                // tensor_dim1[31:16]
        | (32 << 16);                                  // tile_dim0 = 32
  g1[4] = 128;                                         // tile_dim1 = 128
  g1[5] = K;                                           // tensor_dim0_stride
  g1[6] = 0;
  g1[7] = 0;
  i32x4 z4 = {0, 0, 0, 0};
#if __clang_major__ >= 23
  i32x8 z8 = {0, 0, 0, 0, 0, 0, 0, 0};
  __builtin_amdgcn_tensor_load_to_lds(g0, g1, z4, z4, z8, 0);
#else
  __builtin_amdgcn_tensor_load_to_lds(g0, g1, z4, z4, 0);
#endif
}

// ---------------------------------------------------------------------------
// C[M,N] f32 = A[M,K]bf16 * W[N,K]bf16^T + bias[N]
// Block: 256 threads (8 waves, 2x4), tile 128x128, K-step 32.
// Staging: TDM tensor_load_to_lds (double-buffered, issued by wave 0,
// overlapped with WMMA compute; TENSORcnt + workgroup barrier sync).
// ---------------------------------------------------------------------------
__global__ __launch_bounds__(256) void wmma_gemm_bias(
    const bf16* __restrict__ A, const bf16* __restrict__ W,
    const float* __restrict__ bias, float* __restrict__ Cout,
    int M, int N, int K) {
  constexpr int LDSS = 40;  // bf16 elems per LDS row (32 data + 8 pad) = 80B
  __shared__ bf16 As[2][128 * LDSS];
  __shared__ bf16 Bs[2][128 * LDSS];

  const int tid  = threadIdx.x;
  const int lane = tid & 31;
  const int wave = tid >> 5;
  const int wm = wave & 1;        // 2 waves along M
  const int wn = wave >> 1;       // 4 waves along N
  const int lr = lane & 15;       // row/col within a 16x16 tile
  const int hi = lane >> 4;       // lane half selector
  const int blockM = blockIdx.y * 128;
  const int blockN = blockIdx.x * 128;

  const unsigned asb[2] = {(unsigned)(uintptr_t)&As[0][0],
                           (unsigned)(uintptr_t)&As[1][0]};
  const unsigned bsb[2] = {(unsigned)(uintptr_t)&Bs[0][0],
                           (unsigned)(uintptr_t)&Bs[1][0]};

  v8f acc[4][2] = {};
  const int nk = K / 32;

  if (wave == 0) {  // prologue: DMA tile 0 into buffer 0
    tdm_load_tile_128x32(A + (size_t)blockM * K, asb[0], K, M);
    tdm_load_tile_128x32(W + (size_t)blockN * K, bsb[0], K, N);
  }

  for (int kt = 0; kt < nk; ++kt) {
    const int cur = kt & 1;
    if (wave == 0) __builtin_amdgcn_s_wait_tensorcnt(0);  // tile kt landed
    __syncthreads();  // tile kt visible to all; compute kt-1 fully done
    if (wave == 0 && kt + 1 < nk) {  // DMA tile kt+1 into the other buffer
      tdm_load_tile_128x32(A + (size_t)blockM * K + (kt + 1) * 32,
                           asb[cur ^ 1], K, M);
      tdm_load_tile_128x32(W + (size_t)blockN * K + (kt + 1) * 32,
                           bsb[cur ^ 1], K, N);
    }

    const bf16* Asb = As[cur];
    const bf16* Bsb = Bs[cur];

    // ---- B fragments: lane = column n, K {8h..8h+7, 16+8h..16+8h+7} ----
    V16 bfrag[2];
#pragma unroll
    for (int nt = 0; nt < 2; ++nt) {
      int bcol = wn * 32 + nt * 16 + lr;
      int kb   = hi * 8;
      bfrag[nt].h[0] = *(const v8bf*)(&Bsb[bcol * LDSS + kb]);
      bfrag[nt].h[1] = *(const v8bf*)(&Bsb[bcol * LDSS + kb + 16]);
    }
    // ---- A fragments + WMMA ----
#pragma unroll
    for (int mt = 0; mt < 4; ++mt) {
      V16 afrag;
      int arow = wm * 64 + mt * 16 + lr;
      int kb   = hi * 8;
      afrag.h[0] = *(const v8bf*)(&Asb[arow * LDSS + kb]);       // K 0..15
      afrag.h[1] = *(const v8bf*)(&Asb[arow * LDSS + 16 + kb]);  // K 16..31
#pragma unroll
      for (int nt = 0; nt < 2; ++nt) {
        acc[mt][nt] = __builtin_amdgcn_wmma_f32_16x16x32_bf16(
            false, afrag.v, false, bfrag[nt].v, (short)0, acc[mt][nt],
            false, false);
      }
    }
  }

  // ---- epilogue: C layout VGPR r -> M = r + 8*hi, lane&15 -> N ----
#pragma unroll
  for (int mt = 0; mt < 4; ++mt) {
#pragma unroll
    for (int nt = 0; nt < 2; ++nt) {
      int gn = blockN + wn * 32 + nt * 16 + lr;
      float bval = bias[gn];
#pragma unroll
      for (int rr = 0; rr < 8; ++rr) {
        int gm = blockM + wm * 64 + mt * 16 + rr + hi * 8;
        Cout[(size_t)gm * N + gn] = acc[mt][nt][rr] + bval;
      }
    }
  }
}

// ---------------------------------------------------------------------------
// In-place softmax over each consecutive 64-element chunk (one head row).
// One wave32 per chunk; lane handles elements {lane, lane+32}.
// ---------------------------------------------------------------------------
__global__ void softmax64(float* __restrict__ data, int nchunks) {
  int gwarp  = (blockIdx.x * blockDim.x + threadIdx.x) >> 5;
  int nwarps = (gridDim.x * blockDim.x) >> 5;
  int lane   = threadIdx.x & 31;
  for (int c = gwarp; c < nchunks; c += nwarps) {
    float* p = data + (size_t)c * 64;
    float v0 = p[lane], v1 = p[lane + 32];
    float m = fmaxf(v0, v1);
#pragma unroll
    for (int off = 16; off; off >>= 1) m = fmaxf(m, __shfl_xor(m, off, 32));
    float e0 = __expf(v0 - m), e1 = __expf(v1 - m);
    float s = e0 + e1;
#pragma unroll
    for (int off = 16; off; off >>= 1) s += __shfl_xor(s, off, 32);
    float r = 1.0f / s;
    p[lane]      = e0 * r;
    p[lane + 32] = e1 * r;
  }
}

// ---------------------------------------------------------------------------
// kv[b,h,d,e] = sum_t k[b,t,h,d] * v[b,t,h,e]    (D = 64)
// kcum[b,h,d] = sum_t k[b,t,h,d]
// One block per (b,h); 16 T-rows staged in LDS per step.
// ---------------------------------------------------------------------------
__global__ __launch_bounds__(256) void kv_kernel(
    const float* __restrict__ Kd, const float* __restrict__ Vd,
    float* __restrict__ kv, float* __restrict__ kcum, int T, int C) {
  __shared__ float Ks[16 * 64];
  __shared__ float Vs[16 * 64];
  int bh = blockIdx.x;
  int b = bh >> 4, h = bh & 15;
  int tid = threadIdx.x;
  int tr = tid >> 4, q4 = (tid & 15) * 4;  // staging map
  int d  = tid >> 2, eb = (tid & 3) * 16;  // compute map: 16 e's per thread

  size_t base = ((size_t)b * T) * C + h * 64;
  float acc[16] = {};
  float kc = 0.0f;

  for (int t0 = 0; t0 < T; t0 += 16) {
    f32x4 kr = *(const f32x4*)(Kd + base + (size_t)(t0 + tr) * C + q4);
    f32x4 vr = *(const f32x4*)(Vd + base + (size_t)(t0 + tr) * C + q4);
    __syncthreads();
    *(f32x4*)(&Ks[tr * 64 + q4]) = kr;
    *(f32x4*)(&Vs[tr * 64 + q4]) = vr;
    __syncthreads();
    if (tid < 64) {
#pragma unroll
      for (int i = 0; i < 16; ++i) kc += Ks[i * 64 + tid];
    }
#pragma unroll
    for (int i = 0; i < 16; ++i) {
      float kd = Ks[i * 64 + d];
#pragma unroll
      for (int j = 0; j < 16; ++j)
        acc[j] = fmaf(kd, Vs[i * 64 + eb + j], acc[j]);
    }
  }
  float* dst = kv + (size_t)bh * 4096 + d * 64 + eb;
#pragma unroll
  for (int j = 0; j < 16; ++j) dst[j] = acc[j];
  if (tid < 64) kcum[bh * 64 + tid] = kc;
}

// ---------------------------------------------------------------------------
// out[b,t,h,e] = q + (sum_d q[d]*kv[d,e]) / (sum_d q[d]*kcum[d]); bf16 out.
// One block per (b,h, 64 t-rows); kv tile cached in LDS.
// ---------------------------------------------------------------------------
__global__ __launch_bounds__(256) void lin_attn_epilogue(
    const float* __restrict__ Q, const float* __restrict__ kv,
    const float* __restrict__ kcum, bf16* __restrict__ attn, int T, int C) {
  __shared__ float kvs[64 * 64];
  __shared__ float kcs[64];
  __shared__ float Qs[4 * 64];
  int bh = blockIdx.x;
  int b = bh >> 4, h = bh & 15;
  int t0 = blockIdx.y * 64;
  int tid = threadIdx.x;

  const float* kvsrc = kv + (size_t)bh * 4096;
#pragma unroll
  for (int j = 0; j < 4; ++j) {
    int s = tid + j * 256;
    *(f32x4*)(&kvs[s * 4]) = *(const f32x4*)(kvsrc + s * 4);
  }
  if (tid < 64) kcs[tid] = kcum[bh * 64 + tid];
  __syncthreads();

  int r = tid >> 6, e = tid & 63;  // 4 rows x 64 cols per pass
  size_t qbase = ((size_t)b * T) * C + h * 64;
  for (int g = 0; g < 16; ++g) {
    int t = t0 + g * 4 + r;
    float qv = Q[qbase + (size_t)t * C + e];
    Qs[r * 64 + e] = qv;
    __syncthreads();
    float dotk = 0.0f, accv = 0.0f;
#pragma unroll 8
    for (int d = 0; d < 64; ++d) {
      float qd = Qs[r * 64 + d];
      dotk = fmaf(qd, kcs[d], dotk);
      accv = fmaf(qd, kvs[d * 64 + e], accv);
    }
    float o = qv + accv * (1.0f / dotk);
    attn[qbase + (size_t)t * C + e] = (bf16)o;
    __syncthreads();
  }
}

// ---------------------------------------------------------------------------
extern "C" void kernel_launch(void* const* d_in, const int* in_sizes, int n_in,
                              void* d_out, int out_size, void* d_ws, size_t ws_size,
                              hipStream_t stream) {
  (void)in_sizes; (void)n_in; (void)out_size; (void)ws_size;
  const int B = 4, T = 4096, C = 1024, H = 16;
  const int M = B * T;  // 16384 token rows

  const float* x  = (const float*)d_in[0];
  const float* y  = (const float*)d_in[1];
  const float* Wq = (const float*)d_in[2];
  const float* bq = (const float*)d_in[3];
  const float* Wk = (const float*)d_in[4];
  const float* bk = (const float*)d_in[5];
  const float* Wv = (const float*)d_in[6];
  const float* bv = (const float*)d_in[7];
  const float* Wp = (const float*)d_in[8];
  const float* bp = (const float*)d_in[9];
  float* out = (float*)d_out;

  char* ws = (char*)d_ws;
  size_t off = 0;
  auto alloc = [&](size_t bytes) -> void* {
    off = (off + 255) & ~(size_t)255;
    void* p = ws + off;
    off += bytes;
    return p;
  };

  bf16* xb  = (bf16*)alloc((size_t)M * C * sizeof(bf16));
  bf16* yb  = (bf16*)alloc((size_t)M * C * sizeof(bf16));
  bf16* Wqb = (bf16*)alloc((size_t)C * C * sizeof(bf16));
  bf16* Wkb = (bf16*)alloc((size_t)C * C * sizeof(bf16));
  bf16* Wvb = (bf16*)alloc((size_t)C * C * sizeof(bf16));
  bf16* Wpb = (bf16*)alloc((size_t)C * C * sizeof(bf16));
  float* Qf = (float*)alloc((size_t)M * C * sizeof(float));
  float* Kf = (float*)alloc((size_t)M * C * sizeof(float));
  float* Vf = (float*)alloc((size_t)M * C * sizeof(float));
  float* kvb = (float*)alloc((size_t)B * H * 64 * 64 * sizeof(float));
  float* kcb = (float*)alloc((size_t)B * H * 64 * sizeof(float));
  bf16* attnb = (bf16*)Kf;  // K f32 is dead after kv_kernel; reuse its space

  // 1) casts to bf16
  cast_f32_to_bf16<<<2048, 256, 0, stream>>>(x,  xb,  M * C);
  cast_f32_to_bf16<<<2048, 256, 0, stream>>>(y,  yb,  M * C);
  cast_f32_to_bf16<<<512,  256, 0, stream>>>(Wq, Wqb, C * C);
  cast_f32_to_bf16<<<512,  256, 0, stream>>>(Wk, Wkb, C * C);
  cast_f32_to_bf16<<<512,  256, 0, stream>>>(Wv, Wvb, C * C);
  cast_f32_to_bf16<<<512,  256, 0, stream>>>(Wp, Wpb, C * C);

  // 2) Q/K/V projections (WMMA bf16 -> f32, TDM-staged)
  dim3 ggrid(C / 128, M / 128);
  wmma_gemm_bias<<<ggrid, 256, 0, stream>>>(xb, Wqb, bq, Qf, M, C, C);
  wmma_gemm_bias<<<ggrid, 256, 0, stream>>>(yb, Wkb, bk, Kf, M, C, C);
  wmma_gemm_bias<<<ggrid, 256, 0, stream>>>(yb, Wvb, bv, Vf, M, C, C);

  // 3) per-head softmax on Q and K (chunks of 64)
  softmax64<<<2048, 256, 0, stream>>>(Qf, M * H);
  softmax64<<<2048, 256, 0, stream>>>(Kf, M * H);

  // 4) kv = k^T v and k_cumsum per (b,h)
  kv_kernel<<<B * H, 256, 0, stream>>>(Kf, Vf, kvb, kcb, T, C);

  // 5) fused linear-attention epilogue -> bf16 attn output (heads merged)
  lin_attn_epilogue<<<dim3(B * H, T / 64), 256, 0, stream>>>(
      Qf, kvb, kcb, attnb, T, C);

  // 6) output projection (WMMA) -> f32 d_out
  wmma_gemm_bias<<<ggrid, 256, 0, stream>>>(attnb, Wpb, bp, out, M, C, C);
}
